// TMM_44066364457353
// MI455X (gfx1250) — matrix operand order repaired and checked by reading, not compiled, not run on hardware
//
#include <hip/hip_runtime.h>
#include <hip/hip_bf16.h>
#include <math.h>

// ---------------------------------------------------------------------------
// Types for CDNA5 WMMA
// ---------------------------------------------------------------------------
typedef __attribute__((ext_vector_type(16))) __bf16          v16bf;
typedef __attribute__((ext_vector_type(8)))  float           v8f;
typedef __attribute__((ext_vector_type(16))) unsigned short  us16;
typedef __attribute__((ext_vector_type(4)))  unsigned int    u32x4;

// ---------------------------------------------------------------------------
// bf16 helpers (round-to-nearest-even)
// ---------------------------------------------------------------------------
__device__ __forceinline__ unsigned short f32_to_bf16u(float f) {
    unsigned int u = __float_as_uint(f);
    u += 0x7FFFu + ((u >> 16) & 1u);
    return (unsigned short)(u >> 16);
}
__device__ __forceinline__ float bf16u_to_f32(unsigned short h) {
    return __uint_as_float(((unsigned int)h) << 16);
}
__device__ __forceinline__ float sigmoidf_(float x) {
    return 1.0f / (1.0f + __expf(-x));
}
__device__ __forceinline__ float gelu_exact(float x) {
    return 0.5f * x * (1.0f + erff(x * 0.70710678118654752f));
}

// ---------------------------------------------------------------------------
// Problem constants
// ---------------------------------------------------------------------------
#define HD    256          // hidden dim
#define GD    768          // 3*H gate dim
#define SEQS  1024         // B*M sequences
#define TLEN  128          // sequence length
#define ROWS  (SEQS*TLEN)  // 131072 token rows
#define NT    48           // 768/16 n-tiles
#define KS    8            // 256/32 k-steps
// fragment-linear weight layout: [kstep][ntile][lane][16 halfs]
#define FRAG_HALFS 16
#define FRAG_TILE  (32*FRAG_HALFS)          // 512 halfs per (kstep,ntile)
#define KSTEP_HALFS (NT*FRAG_TILE)          // 24576 halfs per kstep
#define LAYER_HALFS (KS*KSTEP_HALFS)        // 196608 halfs per layer

// ---------------------------------------------------------------------------
// A fragment from LDS (row-major [m][k], 16-bit), ISA 7.12.2 layout:
// lanes0-15: M=lane, V0-3 K=0..7, V4-7 K=16..23; lanes16-31: K offset +8.
// Lowers to 2x ds_load_b128 per wave.
// ---------------------------------------------------------------------------
__device__ __forceinline__ v16bf load_a_frag(const unsigned short* src,
                                             int row_stride, int kbase, int lane) {
    int m = lane & 15;
    int g = (lane >> 4) << 3;   // 0 or 8
    us16 t;
#pragma unroll
    for (int v = 0; v < 8; ++v) {
        int k = ((v < 4) ? (2 * v) : (16 + 2 * (v - 4))) + g + kbase;
        t[2 * v]     = src[m * row_stride + k];
        t[2 * v + 1] = src[m * row_stride + k + 1];
    }
    union { us16 u; v16bf b; } cvt; cvt.u = t; return cvt.b;
}

// B fragment: per-thread base pointer + compile-time half offset ->
// global_load_b128 x2 with immediate IOFFSET, no per-fragment address math.
__device__ __forceinline__ v16bf load_b_off(const unsigned short* __restrict__ base,
                                            int half_off) {
    union { us16 u; v16bf b; } cvt;
    cvt.u = *(const us16*)(base + half_off);
    return cvt.b;
}

// D/C 16x16 f32: VGPR r -> M = r + 8*(lane>=16), N = lane&15
__device__ __forceinline__ void store_acc_lds(float* dst, int row_stride,
                                              int colbase, int lane, v8f acc) {
    int n  = (lane & 15) + colbase;
    int mo = (lane >> 4) << 3;
#pragma unroll
    for (int r = 0; r < 8; ++r) dst[(mo + r) * row_stride + n] = acc[r];
}

// ===========================================================================
// Kernel 1: fp32 weights -> bf16 in WMMA-fragment-linear layout.
//   dst i (per layer): j = i&15, lane = (i>>4)&31, idx = i>>9,
//   ntile = idx%NT, kstep = idx/NT
//   k = kstep*32 + (lane>>4)*8 + (j&7) + (j>>3)*16,  g = ntile*16 + (lane&15)
//   src = W[l][g][k]  (W row-major [l][768][256])
// ===========================================================================
__global__ void prep_weights_kernel(const float* __restrict__ Wih,
                                    const float* __restrict__ Whh,
                                    unsigned short* __restrict__ WihF,
                                    unsigned short* __restrict__ WhhF) {
    int i = blockIdx.x * blockDim.x + threadIdx.x;
    if (i >= 3 * LAYER_HALFS) return;
    int l = i / LAYER_HALFS;
    int r = i - l * LAYER_HALFS;
    int j    = r & 15;
    int lane = (r >> 4) & 31;
    int idx  = r >> 9;
    int nt   = idx % NT;
    int ks   = idx / NT;
    int k = ks * 32 + ((lane >> 4) << 3) + (j & 7) + ((j >> 3) << 4);
    int g = nt * 16 + (lane & 15);
    size_t src = (size_t)l * (GD * HD) + (size_t)g * HD + k;
    WihF[i] = f32_to_bf16u(Wih[src]);
    WhhF[i] = f32_to_bf16u(Whh[src]);
}

// ===========================================================================
// Kernel 2: x fp32 -> bf16 sequence buffer (layer-0 input)
// ===========================================================================
__global__ void x_to_bf16_kernel(const float* __restrict__ x,
                                 unsigned short* __restrict__ seq) {
    int base = (blockIdx.x * blockDim.x + threadIdx.x) * 4;
#pragma unroll
    for (int q = 0; q < 4; ++q) seq[base + q] = f32_to_bf16u(x[base + q]);
}

// ===========================================================================
// Kernel 3: xg = seqIn @ Wih^T + bih, stored bf16 [ROWS][768]
// Block tile 64(M) x 128(N); 256 threads = 8 waves; wave = 16x64 (4 N-tiles).
// A staged in LDS; B via one base pointer + immediate offsets.
// ===========================================================================
#define ASTR 40
#define GSTR 132

__global__ void xg_gemm_kernel(const unsigned short* __restrict__ seqIn,
                               const unsigned short* __restrict__ WihF,
                               const float* __restrict__ bih,
                               unsigned short* __restrict__ xg) {
    extern __shared__ char smem[];
    float*          gst = (float*)smem;                       // 64*132*4 = 33792
    unsigned short* ast = (unsigned short*)(smem + 33792);    // 64*40*2  = 5120

    const int tid  = threadIdx.x;
    const int lane = tid & 31;
    const int wid  = tid >> 5;
    const int wm   = wid >> 1;            // 0..3 : M sub-tile
    const int wn   = wid & 1;             // 0..1 : N quad
    const int rowbase = blockIdx.x * 64;
    const int ntbase  = blockIdx.y * 8;   // first of 8 n-tiles

    // per-thread weight base: covers (ntbase + wn*4, lane); fragment (ks,j)
    // sits at constant half offset ks*KSTEP_HALFS + j*FRAG_TILE.
    const unsigned short* wbase =
        WihF + (size_t)(ntbase + wn * 4) * FRAG_TILE + lane * FRAG_HALFS;

    v8f acc[4];
#pragma unroll
    for (int j = 0; j < 4; ++j)
#pragma unroll
        for (int q = 0; q < 8; ++q) acc[j][q] = 0.0f;

#pragma unroll
    for (int ks = 0; ks < KS; ++ks) {
        const int kbase = ks * 32;
        __syncthreads();
        // stage A: 64 rows x 32 halfs (one uint4 per thread, coalesced)
        {
            int r = tid >> 2, c = (tid & 3) * 8;
            *(u32x4*)(ast + r * ASTR + c) =
                *(const u32x4*)(seqIn + (size_t)(rowbase + r) * HD + kbase + c);
        }
        __syncthreads();

        v16bf a = load_a_frag(ast + wm * 16 * ASTR, ASTR, 0, lane);
#pragma unroll
        for (int j = 0; j < 4; ++j) {
            v16bf b = load_b_off(wbase, ks * KSTEP_HALFS + j * FRAG_TILE);
            acc[j] = __builtin_amdgcn_wmma_f32_16x16x32_bf16(
                false, a, false, b, (short)0, acc[j], false, false);
        }
    }

#pragma unroll
    for (int j = 0; j < 4; ++j)
        store_acc_lds(gst + wm * 16 * GSTR, GSTR, (wn * 4 + j) * 16, lane, acc[j]);
    __syncthreads();

    // epilogue: +bias, f32 -> bf16, packed 32-bit coalesced stores
    const int colbase = ntbase * 16;
#pragma unroll 4
    for (int q = 0; q < 16; ++q) {
        int e   = q * 256 + tid;       // uint index in 64x64 uints
        int row = e >> 6;
        int cu  = e & 63;
        float v0 = gst[row * GSTR + 2 * cu]     + bih[colbase + 2 * cu];
        float v1 = gst[row * GSTR + 2 * cu + 1] + bih[colbase + 2 * cu + 1];
        unsigned int pk = (unsigned int)f32_to_bf16u(v0) |
                          ((unsigned int)f32_to_bf16u(v1) << 16);
        *(unsigned int*)(xg + (size_t)(rowbase + row) * GD + colbase + 2 * cu) = pk;
    }
}

// ===========================================================================
// Kernel 4: fused GRU recurrence, one layer.
// 64 blocks x 16 sequences; 256 threads = 8 waves; per timestep each wave
// computes 6 16x16 tiles of hg = h @ Whh^T (K=256 -> 8 x wmma_bf16).
// h lives in LDS (bf16); Whh via one base pointer + immediate offsets.
// ===========================================================================
#define HSTR  272
#define HGSTR 776

__global__ void gru_rec_kernel(const unsigned short* __restrict__ xg,
                               const unsigned short* __restrict__ WhhF,
                               const float* __restrict__ bhh,
                               unsigned short* __restrict__ seqOut) {
    extern __shared__ char smem[];
    float*          hg   = (float*)smem;                      // 16*776*4 = 49664
    unsigned short* hbuf = (unsigned short*)(smem + 49664);   // 16*272*2 = 8704

    const int tid  = threadIdx.x;
    const int lane = tid & 31;
    const int wid  = tid >> 5;
    const int sb   = blockIdx.x * 16;

    // per-thread weight base: covers (ntile = wid + 8j via offset, lane);
    // fragment (ks,j) at constant half offset ks*KSTEP_HALFS + j*8*FRAG_TILE.
    const unsigned short* wbase =
        WhhF + (size_t)wid * FRAG_TILE + lane * FRAG_HALFS;

    // h0 = 0
    for (int i = tid; i < 16 * HSTR; i += 256) hbuf[i] = 0;

    const float brr = bhh[tid];
    const float bzz = bhh[256 + tid];
    const float bnn = bhh[512 + tid];

    for (int t = 0; t < TLEN; ++t) {
        v8f acc[6];
#pragma unroll
        for (int j = 0; j < 6; ++j)
#pragma unroll
            for (int q = 0; q < 8; ++q) acc[j][q] = 0.0f;

        __syncthreads();   // h(t) in hbuf is visible to all waves

#pragma unroll
        for (int ks = 0; ks < KS; ++ks) {
            v16bf a = load_a_frag(hbuf, HSTR, ks * 32, lane);
#pragma unroll
            for (int j = 0; j < 6; ++j) {
                v16bf b = load_b_off(wbase, ks * KSTEP_HALFS + j * 8 * FRAG_TILE);
                acc[j] = __builtin_amdgcn_wmma_f32_16x16x32_bf16(
                    false, a, false, b, (short)0, acc[j], false, false);
            }
        }

#pragma unroll
        for (int j = 0; j < 6; ++j)
            store_acc_lds(hg, HGSTR, (wid + 8 * j) * 16, lane, acc[j]);
        __syncthreads();

        // elementwise GRU update: thread tid handles hidden index i=tid for
        // each of the 16 sequences in the tile.
        for (int m = 0; m < 16; ++m) {
            const size_t row = ((size_t)(sb + m) * TLEN + t) * GD;
            float xr = bf16u_to_f32(xg[row + tid]);
            float xz = bf16u_to_f32(xg[row + 256 + tid]);
            float xn = bf16u_to_f32(xg[row + 512 + tid]);
            float hr = hg[m * HGSTR + tid]       + brr;
            float hz = hg[m * HGSTR + 256 + tid] + bzz;
            float hn = hg[m * HGSTR + 512 + tid] + bnn;
            float r = sigmoidf_(xr + hr);
            float z = sigmoidf_(xz + hz);
            float n = tanhf(xn + r * hn);
            float hp = bf16u_to_f32(hbuf[m * HSTR + tid]);
            float hnew = (1.0f - z) * n + z * hp;
            unsigned short hb = f32_to_bf16u(hnew);
            hbuf[m * HSTR + tid] = hb;
            seqOut[((size_t)(sb + m) * TLEN + t) * HD + tid] = hb;
        }
        // next iteration's leading __syncthreads() orders hbuf/hg reuse
    }
}

// ===========================================================================
// Kernel 5: MLP head on last timestep. 64 blocks x 16 sequences, VALU.
// ===========================================================================
__global__ void mlp_head_kernel(const unsigned short* __restrict__ seq,
                                const float* __restrict__ W1, const float* __restrict__ b1,
                                const float* __restrict__ W2, const float* __restrict__ b2,
                                const float* __restrict__ W3, const float* __restrict__ b3,
                                float* __restrict__ out) {
    extern __shared__ char smem[];
    float* last = (float*)smem;        // 16 x 264
    float* h1   = last + 16 * 264;
    float* h2   = h1 + 16 * 264;
    float* red  = h2 + 16 * 264;       // 256
    const int tid = threadIdx.x;
    const int sb  = blockIdx.x * 16;

    for (int m = 0; m < 16; ++m)
        last[m * 264 + tid] =
            bf16u_to_f32(seq[((size_t)(sb + m) * TLEN + (TLEN - 1)) * HD + tid]);
    __syncthreads();

    for (int m = 0; m < 16; ++m) {
        float a = b1[tid];
        const float* w = W1 + (size_t)tid * HD;
        for (int k = 0; k < HD; ++k) a = fmaf(last[m * 264 + k], w[k], a);
        h1[m * 264 + tid] = gelu_exact(a);
    }
    __syncthreads();
    for (int m = 0; m < 16; ++m) {
        float a = b2[tid];
        const float* w = W2 + (size_t)tid * HD;
        for (int k = 0; k < HD; ++k) a = fmaf(h1[m * 264 + k], w[k], a);
        h2[m * 264 + tid] = gelu_exact(a);
    }
    __syncthreads();
    for (int m = 0; m < 16; ++m) {
        red[tid] = h2[m * 264 + tid] * W3[tid];
        __syncthreads();
        for (int off = 128; off > 0; off >>= 1) {
            if (tid < off) red[tid] += red[tid + off];
            __syncthreads();
        }
        if (tid == 0) out[sb + m] = red[0] + b3[0];
        __syncthreads();
    }
}

// ===========================================================================
// Host launcher
// ===========================================================================
extern "C" void kernel_launch(void* const* d_in, const int* in_sizes, int n_in,
                              void* d_out, int out_size, void* d_ws, size_t ws_size,
                              hipStream_t stream) {
    (void)in_sizes; (void)n_in; (void)out_size; (void)ws_size;

    const float* x   = (const float*)d_in[0];
    const float* Wih = (const float*)d_in[1];
    const float* Whh = (const float*)d_in[2];
    const float* bih = (const float*)d_in[3];
    const float* bhh = (const float*)d_in[4];
    const float* W1  = (const float*)d_in[5];
    const float* b1  = (const float*)d_in[6];
    const float* W2  = (const float*)d_in[7];
    const float* b2  = (const float*)d_in[8];
    const float* W3  = (const float*)d_in[9];
    const float* b3  = (const float*)d_in[10];
    float* out = (float*)d_out;

    char* ws = (char*)d_ws;
    unsigned short* WihF = (unsigned short*)ws;                      // 1,179,648 B
    unsigned short* WhhF = (unsigned short*)(ws + 1179648);          // 1,179,648 B
    unsigned short* seqA = (unsigned short*)(ws + 2359296);          // 67,108,864 B
    unsigned short* seqB = (unsigned short*)(ws + 69468160);         // 67,108,864 B
    unsigned short* xg   = (unsigned short*)(ws + 136577024);        // 201,326,592 B

    // 1) weights -> bf16 fragment-linear
    {
        int total = 3 * LAYER_HALFS;
        prep_weights_kernel<<<(total + 255) / 256, 256, 0, stream>>>(Wih, Whh, WihF, WhhF);
    }
    // 2) x -> bf16
    x_to_bf16_kernel<<<(ROWS * HD) / (256 * 4), 256, 0, stream>>>(x, seqA);

    const size_t gemm_smem = 33792 + 5120;          // ~38 KB
    const size_t gru_smem  = 49664 + 8704;          // ~57 KB
    const size_t mlp_smem  = (3 * 16 * 264 + 256) * sizeof(float);

    for (int l = 0; l < 3; ++l) {
        const unsigned short* sin  = (l & 1) ? seqB : seqA;
        unsigned short*       sout = (l & 1) ? seqA : seqB;
        const unsigned short* wihf = WihF + (size_t)l * LAYER_HALFS;
        const unsigned short* whhf = WhhF + (size_t)l * LAYER_HALFS;

        // 3) input-side gates for all timesteps (full-occupancy WMMA GEMM)
        xg_gemm_kernel<<<dim3(ROWS / 64, NT / 8), 256, gemm_smem, stream>>>(
            sin, wihf, bih + l * GD, xg);

        // 4) fused recurrence
        gru_rec_kernel<<<SEQS / 16, 256, gru_smem, stream>>>(
            xg, whhf, bhh + l * GD, sout);
    }

    // 5) MLP head (layer 2 wrote seqB)
    mlp_head_kernel<<<SEQS / 16, 256, mlp_smem, stream>>>(
        seqB, W1, b1, W2, b2, W3, b3, out);
}
